// multihead_attn_17325898072207
// MI455X (gfx1250) — matrix operand order repaired and checked by reading, not compiled, not run on hardware
//
#include <hip/hip_runtime.h>

// ---------------------------------------------------------------------------
// Types for CDNA5 WMMA (wave32)
// ---------------------------------------------------------------------------
typedef __bf16 bf16_t;
typedef __attribute__((ext_vector_type(16))) __bf16 v16bf;
typedef __attribute__((ext_vector_type(8)))  __bf16 v8bf;   // 16 bytes
typedef __attribute__((ext_vector_type(4)))  __bf16 v4bf;   // 8 bytes
typedef __attribute__((ext_vector_type(8)))  float  v8f;

union FragBF { v16bf v; v8bf h[2]; };

#define WMMA_BF16(A, B, C) \
  __builtin_amdgcn_wmma_f32_16x16x32_bf16(false, (A), false, (B), (short)0, (C), false, false)

#define D_MODEL 1024
#define N_HEADS 16
#define HEAD_DIM 64
#define SEQ 2048
#define BATCH 2
#define M_TOTAL (BATCH * SEQ)   // 4096

// ---------------------------------------------------------------------------
// Async global -> LDS copy (CDNA5, ASYNCcnt). One 16-byte chunk per lane.
// LDS address operand = low 32 bits of the generic pointer.
// ---------------------------------------------------------------------------
__device__ __forceinline__ void async_copy_b128(void* lds_dst, const void* gsrc) {
  unsigned loff = (unsigned)(size_t)lds_dst;
  unsigned long long ga = (unsigned long long)(size_t)gsrc;
  asm volatile("global_load_async_to_lds_b128 %0, %1, off"
               :: "v"(loff), "v"(ga) : "memory");
}
// ASYNCcnt decrements in issue order -> partial waits give double buffering.
__device__ __forceinline__ void wait_async_le0() {
  asm volatile("s_wait_asynccnt 0x0" ::: "memory");
}
__device__ __forceinline__ void wait_async_le4() {
  asm volatile("s_wait_asynccnt 0x4" ::: "memory");
}
__device__ __forceinline__ void wait_async_le8() {
  asm volatile("s_wait_asynccnt 0x8" ::: "memory");
}

// ---------------------------------------------------------------------------
// Kernel 1a: f32 -> bf16 conversion (vectorized x4), for activations x
// ---------------------------------------------------------------------------
__global__ __launch_bounds__(256) void cvt_f32_to_bf16(const float* __restrict__ in,
                                                       bf16_t* __restrict__ out, int n4) {
  int i = blockIdx.x * blockDim.x + threadIdx.x;
  if (i < n4) {
    float4 f = ((const float4*)in)[i];
    v4bf o;
    o[0] = (bf16_t)f.x; o[1] = (bf16_t)f.y; o[2] = (bf16_t)f.z; o[3] = (bf16_t)f.w;
    ((v4bf*)out)[i] = o;
  }
}

// ---------------------------------------------------------------------------
// Kernel 1b: weight transpose + convert:  W[k][n] f32  ->  Wt[n][k] bf16.
// ---------------------------------------------------------------------------
__global__ __launch_bounds__(256) void transpose_cvt_kernel(const float* __restrict__ W,
                                                            bf16_t* __restrict__ Wt) {
  __shared__ bf16_t tile[32][33];
  const int k0 = blockIdx.x * 32, n0 = blockIdx.y * 32;
  const int rr = threadIdx.x >> 5, c = threadIdx.x & 31;
#pragma unroll
  for (int i = 0; i < 4; ++i) {
    int r = i * 8 + rr;
    tile[r][c] = (bf16_t)W[(size_t)(k0 + r) * D_MODEL + n0 + c];
  }
  __syncthreads();
#pragma unroll
  for (int i = 0; i < 4; ++i) {
    int r = i * 8 + rr;
    Wt[(size_t)(n0 + r) * D_MODEL + k0 + c] = tile[c][r];
  }
}

// ---------------------------------------------------------------------------
// Async-staged, double-buffered GEMM mainloop:
//   C(128x128) += A(128xK) * Bt(128xK)^T, bf16 in, f32 acc.
// BK=32 per stage, LDS row stride 40 bf16 (80B). Each thread issues 4 async
// b128 copies per stage (2 for A, 2 for Bt); with the next stage in flight,
// s_wait_asynccnt 4 completes the current one (in-order completion).
// LDS: 2 buffers x (A + B) x 10240 B = 40 KB.
// ---------------------------------------------------------------------------
#define GSTRIDE 40
#define GBUF (128 * GSTRIDE)

__device__ __forceinline__ void gemm_stage_async(const bf16_t* __restrict__ A,
                                                 const bf16_t* __restrict__ Bt,
                                                 bf16_t* As, bf16_t* Bs,
                                                 int m0, int n0, int k0, int tid) {
  const int s = tid & 3;        // 8-elem segment (32 cols = 4 segs)
  const int rb = tid >> 2;      // 0..63
#pragma unroll
  for (int i = 0; i < 2; ++i) {
    int r = rb + i * 64;        // 0..127
    async_copy_b128(&As[r * GSTRIDE + s * 8],
                    &A[(size_t)(m0 + r) * D_MODEL + k0 + s * 8]);
    async_copy_b128(&Bs[r * GSTRIDE + s * 8],
                    &Bt[(size_t)(n0 + r) * D_MODEL + k0 + s * 8]);
    if (k0 + 64 < D_MODEL) {    // warm GL2 two stages further ahead
      __builtin_prefetch(&A[(size_t)(m0 + r) * D_MODEL + k0 + 64 + s * 8], 0, 1);
      __builtin_prefetch(&Bt[(size_t)(n0 + r) * D_MODEL + k0 + 64 + s * 8], 0, 1);
    }
  }
}

__device__ __forceinline__ void gemm_mainloop(const bf16_t* __restrict__ A,
                                              const bf16_t* __restrict__ Bt,
                                              bf16_t* As, bf16_t* Bs,  // [2*GBUF] each
                                              int m0, int n0, v8f acc[2][4]) {
  const int tid  = threadIdx.x;
  const int lane = tid & 31;
  const int wave = tid >> 5;
  const int lm = lane & 15;
  const int hf = lane >> 4;
  const int wm = wave >> 1;   // 0..3 along M
  const int wn = wave & 1;    // 0..1 along N
  const int NSTAGE = D_MODEL / 32;   // 32

  gemm_stage_async(A, Bt, As, Bs, m0, n0, 0, tid);

  for (int kk = 0; kk < NSTAGE; ++kk) {
    if (kk + 1 < NSTAGE) {
      int nb = (kk + 1) & 1;
      gemm_stage_async(A, Bt, As + nb * GBUF, Bs + nb * GBUF,
                       m0, n0, (kk + 1) * 32, tid);
      wait_async_le4();        // current stage complete, next stays in flight
    } else {
      wait_async_le0();
    }
    __syncthreads();

    const bf16_t* Ac = As + (kk & 1) * GBUF;
    const bf16_t* Bc = Bs + (kk & 1) * GBUF;
    FragBF a[2], b[4];
#pragma unroll
    for (int i = 0; i < 2; ++i) {
      int row = wm * 32 + i * 16 + lm;
      a[i].h[0] = *(const v8bf*)&Ac[row * GSTRIDE + hf * 8];
      a[i].h[1] = *(const v8bf*)&Ac[row * GSTRIDE + 16 + hf * 8];
    }
#pragma unroll
    for (int j = 0; j < 4; ++j) {
      int col = wn * 64 + j * 16 + lm;
      b[j].h[0] = *(const v8bf*)&Bc[col * GSTRIDE + hf * 8];
      b[j].h[1] = *(const v8bf*)&Bc[col * GSTRIDE + 16 + hf * 8];
    }
#pragma unroll
    for (int i = 0; i < 2; ++i)
#pragma unroll
      for (int j = 0; j < 4; ++j)
        acc[i][j] = WMMA_BF16(a[i].v, b[j].v, acc[i][j]);
    __syncthreads();   // protect buffer before it is re-issued next iteration
  }
}

// ---------------------------------------------------------------------------
// Kernel 2: fused QKV projection. grid = (32, 8, 3); z selects Wq/Wk/Wv.
// Q,K written [B,H,S,hd]; V written pre-transposed [B,H,hd,S].
// ---------------------------------------------------------------------------
__global__ __launch_bounds__(256) void gemm_qkv_kernel(
    const bf16_t* __restrict__ X,
    const bf16_t* __restrict__ Wqt, const bf16_t* __restrict__ Wkt,
    const bf16_t* __restrict__ Wvt,
    bf16_t* __restrict__ Q, bf16_t* __restrict__ K, bf16_t* __restrict__ Vt) {
  __shared__ __align__(16) bf16_t As[2 * GBUF];
  __shared__ __align__(16) bf16_t Bs[2 * GBUF];
  const int z = blockIdx.z;
  const bf16_t* W = (z == 0) ? Wqt : (z == 1) ? Wkt : Wvt;

  const int m0 = blockIdx.x * 128, n0 = blockIdx.y * 128;
  v8f acc[2][4];
  const v8f vzero = {0.f, 0.f, 0.f, 0.f, 0.f, 0.f, 0.f, 0.f};
#pragma unroll
  for (int i = 0; i < 2; ++i)
#pragma unroll
    for (int j = 0; j < 4; ++j) acc[i][j] = vzero;

  gemm_mainloop(X, W, As, Bs, m0, n0, acc);

  const int lane = threadIdx.x & 31, wave = threadIdx.x >> 5;
  const int lm = lane & 15, hf = lane >> 4;
  const int wm = wave >> 1, wn = wave & 1;

  if (z == 2) {                         // V -> [B,H,hd,S]
#pragma unroll
    for (int i = 0; i < 2; ++i)
#pragma unroll
      for (int j = 0; j < 4; ++j)
#pragma unroll
        for (int r = 0; r < 8; ++r) {
          int row = m0 + wm * 32 + i * 16 + r + hf * 8;
          int col = n0 + wn * 64 + j * 16 + lm;
          int bb = row >> 11, ss = row & (SEQ - 1);
          int hh = col >> 6, dd = col & 63;
          Vt[((size_t)(bb * N_HEADS + hh) * HEAD_DIM + dd) * SEQ + ss] =
              (bf16_t)acc[i][j][r];
        }
  } else {                              // Q / K -> [B,H,S,hd]
    bf16_t* O = (z == 0) ? Q : K;
#pragma unroll
    for (int i = 0; i < 2; ++i)
#pragma unroll
      for (int j = 0; j < 4; ++j)
#pragma unroll
        for (int r = 0; r < 8; ++r) {
          int row = m0 + wm * 32 + i * 16 + r + hf * 8;
          int col = n0 + wn * 64 + j * 16 + lm;
          int bb = row >> 11, ss = row & (SEQ - 1);
          int hh = col >> 6, dd = col & 63;
          O[(((size_t)(bb * N_HEADS + hh) * SEQ + ss) * HEAD_DIM) + dd] =
              (bf16_t)acc[i][j][r];
        }
  }
}

// ---------------------------------------------------------------------------
// Kernel 3: causal flash attention, one block per (b*h, 64-query tile).
// 128 threads = 4 waves. K/V tiles double-buffered via async-to-LDS: tile
// j+1 is issued (8 asyncs/thread) before s_wait_asynccnt 8 completes tile j,
// so HBM latency hides behind the WMMAs.
// ---------------------------------------------------------------------------
__global__ __launch_bounds__(128) void flash_attn_kernel(
    const bf16_t* __restrict__ Q, const bf16_t* __restrict__ Kg,
    const bf16_t* __restrict__ VtG, bf16_t* __restrict__ ctx) {
  __shared__ __align__(16) bf16_t Ks [2][64 * 72];   // K tile   [key][d]
  __shared__ __align__(16) bf16_t Vts[2][64 * 72];   // Vt tile  [d][key]
  __shared__ __align__(16) bf16_t Ps [4 * 16 * 72];  // per-wave P (16 x 64)

  const int bh = blockIdx.y;             // b*16 + h
  const int q0 = blockIdx.x * 64;
  const int tid = threadIdx.x;
  const int wave = tid >> 5, lane = tid & 31;
  const int lm = lane & 15, hf = lane >> 4;
  const size_t base = (size_t)bh * SEQ * HEAD_DIM;   // == bh * HEAD_DIM * SEQ

  const int rb = tid >> 3, sseg = tid & 7;
  auto stage = [&](int j0, int buf) {
#pragma unroll
    for (int i = 0; i < 4; ++i) {
      int r = rb + i * 16;               // 0..63
      async_copy_b128(&Ks [buf][r * 72 + sseg * 8],
                      &Kg [base + (size_t)(j0 + r) * HEAD_DIM + sseg * 8]);
      async_copy_b128(&Vts[buf][r * 72 + sseg * 8],
                      &VtG[base + (size_t)r * SEQ + j0 + sseg * 8]);
    }
  };

  // Q fragments (A-matrix, 16x32 per k-step), straight from global.
  FragBF aq[2];
  const int qrow = q0 + wave * 16 + lm;
  const bf16_t* qp = Q + base + (size_t)qrow * HEAD_DIM;
#pragma unroll
  for (int ks = 0; ks < 2; ++ks) {
    aq[ks].h[0] = *(const v8bf*)(qp + ks * 32 + hf * 8);
    aq[ks].h[1] = *(const v8bf*)(qp + ks * 32 + 16 + hf * 8);
  }

  float m_run[8], l_run[8];
  v8f accc[4];
  const v8f vzero = {0.f, 0.f, 0.f, 0.f, 0.f, 0.f, 0.f, 0.f};
#pragma unroll
  for (int r = 0; r < 8; ++r) { m_run[r] = -1e30f; l_run[r] = 0.f; }
#pragma unroll
  for (int t = 0; t < 4; ++t) accc[t] = vzero;

  const int rowg = q0 + wave * 16 + hf * 8;   // query row of acc element r==0
  const int nj = blockIdx.x + 1;              // causal: tiles 0 .. q0/64

  stage(0, 0);
  for (int jj = 0; jj < nj; ++jj) {
    const int j0 = jj * 64;
    if (jj + 1 < nj) {
      stage(j0 + 64, (jj + 1) & 1);
      wait_async_le8();                  // tile jj done; tile jj+1 in flight
    } else {
      wait_async_le0();
    }
    __syncthreads();
    const bf16_t* Kc = Ks [jj & 1];
    const bf16_t* Vc = Vts[jj & 1];

    // ---- scores S = Q * K^T (16 x 64), scaled ----
    float sc[4][8];
#pragma unroll
    for (int t = 0; t < 4; ++t) {
      v8f s8 = vzero;
#pragma unroll
      for (int ks = 0; ks < 2; ++ks) {
        FragBF bk;
        const int kr = t * 16 + lm;      // key row = B column
        bk.h[0] = *(const v8bf*)&Kc[kr * 72 + ks * 32 + hf * 8];
        bk.h[1] = *(const v8bf*)&Kc[kr * 72 + ks * 32 + 16 + hf * 8];
        s8 = WMMA_BF16(aq[ks].v, bk.v, s8);
      }
#pragma unroll
      for (int r = 0; r < 8; ++r) {
        float s = s8[r] * 0.125f;        // 1/sqrt(64)
        if (j0 + t * 16 + lm > rowg + r) s = -1e30f;   // causal mask
        sc[t][r] = s;
      }
    }

    // ---- online softmax (rows split across lane halves; width-16 shuffles) ----
#pragma unroll
    for (int r = 0; r < 8; ++r) {
      float mx = fmaxf(fmaxf(sc[0][r], sc[1][r]), fmaxf(sc[2][r], sc[3][r]));
#pragma unroll
      for (int off = 8; off > 0; off >>= 1) mx = fmaxf(mx, __shfl_xor(mx, off, 16));
      float mnew = fmaxf(m_run[r], mx);
      float corr = __expf(m_run[r] - mnew);
      m_run[r] = mnew;
      float rs = 0.f;
#pragma unroll
      for (int t = 0; t < 4; ++t) {
        float p = __expf(sc[t][r] - mnew);
        sc[t][r] = p;
        rs += p;
      }
#pragma unroll
      for (int off = 8; off > 0; off >>= 1) rs += __shfl_xor(rs, off, 16);
      l_run[r] = l_run[r] * corr + rs;
#pragma unroll
      for (int t = 0; t < 4; ++t) accc[t][r] *= corr;
      // write P (C layout -> LDS row-major) for A-fragment reload
#pragma unroll
      for (int t = 0; t < 4; ++t)
        Ps[wave * 16 * 72 + (r + hf * 8) * 72 + t * 16 + lm] = (bf16_t)sc[t][r];
    }
    // intra-wave LDS store->load ordering for the P round-trip
    asm volatile("s_wait_dscnt 0" ::: "memory");

    // ---- ctx += P * V ----
    const bf16_t* pw = &Ps[wave * 16 * 72];
#pragma unroll
    for (int ks = 0; ks < 2; ++ks) {
      FragBF ap;
      ap.h[0] = *(const v8bf*)&pw[lm * 72 + ks * 32 + hf * 8];
      ap.h[1] = *(const v8bf*)&pw[lm * 72 + ks * 32 + 16 + hf * 8];
#pragma unroll
      for (int td = 0; td < 4; ++td) {
        FragBF bv;
        const int dc = td * 16 + lm;     // output d = B column
        bv.h[0] = *(const v8bf*)&Vc[dc * 72 + ks * 32 + hf * 8];
        bv.h[1] = *(const v8bf*)&Vc[dc * 72 + ks * 32 + 16 + hf * 8];
        accc[td] = WMMA_BF16(ap.v, bv.v, accc[td]);
      }
    }
    __syncthreads();   // all waves done with buffer jj&1 before re-issue
  }

  // ---- normalize and store ctx in [B, S, H*hd] bf16 layout ----
  const int bb = bh >> 4, hh = bh & 15;
#pragma unroll
  for (int r = 0; r < 8; ++r) {
    int row = q0 + wave * 16 + r + hf * 8;
    float inv = 1.0f / l_run[r];
#pragma unroll
    for (int td = 0; td < 4; ++td)
      ctx[((size_t)(bb * SEQ + row)) * D_MODEL + hh * HEAD_DIM + td * 16 + lm] =
          (bf16_t)(accc[td][r] * inv);
  }
}

// ---------------------------------------------------------------------------
// Kernel 4: output projection  out = ctx @ Wo + bo   (f32 result)
// ---------------------------------------------------------------------------
__global__ __launch_bounds__(256) void gemm_out_kernel(
    const bf16_t* __restrict__ C, const bf16_t* __restrict__ Wot,
    const float* __restrict__ bo, float* __restrict__ out) {
  __shared__ __align__(16) bf16_t As[2 * GBUF];
  __shared__ __align__(16) bf16_t Bs[2 * GBUF];
  const int m0 = blockIdx.x * 128, n0 = blockIdx.y * 128;
  v8f acc[2][4];
  const v8f vzero = {0.f, 0.f, 0.f, 0.f, 0.f, 0.f, 0.f, 0.f};
#pragma unroll
  for (int i = 0; i < 2; ++i)
#pragma unroll
    for (int j = 0; j < 4; ++j) acc[i][j] = vzero;

  gemm_mainloop(C, Wot, As, Bs, m0, n0, acc);

  const int lane = threadIdx.x & 31, wave = threadIdx.x >> 5;
  const int lm = lane & 15, hf = lane >> 4;
  const int wm = wave >> 1, wn = wave & 1;
#pragma unroll
  for (int i = 0; i < 2; ++i)
#pragma unroll
    for (int j = 0; j < 4; ++j) {
      int col = n0 + wn * 64 + j * 16 + lm;
      float bias = bo[col];
#pragma unroll
      for (int r = 0; r < 8; ++r) {
        int row = m0 + wm * 32 + i * 16 + r + hf * 8;
        out[(size_t)row * D_MODEL + col] = acc[i][j][r] + bias;
      }
    }
}

// ---------------------------------------------------------------------------
// Host launcher
// ---------------------------------------------------------------------------
extern "C" void kernel_launch(void* const* d_in, const int* in_sizes, int n_in,
                              void* d_out, int out_size, void* d_ws, size_t ws_size,
                              hipStream_t stream) {
  const float* x  = (const float*)d_in[0];
  const float* Wq = (const float*)d_in[1];
  const float* Wk = (const float*)d_in[2];
  const float* Wv = (const float*)d_in[3];
  const float* Wo = (const float*)d_in[4];
  const float* bo = (const float*)d_in[5];

  // Workspace layout (bf16 elements), total ~48 MB
  bf16_t* ws   = (bf16_t*)d_ws;
  bf16_t* xb   = ws;                                   // 4096*1024
  bf16_t* wqt  = xb  + (size_t)M_TOTAL * D_MODEL;      // transposed weights
  bf16_t* wkt  = wqt + (size_t)D_MODEL * D_MODEL;
  bf16_t* wvt  = wkt + (size_t)D_MODEL * D_MODEL;
  bf16_t* wot  = wvt + (size_t)D_MODEL * D_MODEL;
  bf16_t* Qb   = wot + (size_t)D_MODEL * D_MODEL;      // [B,H,S,hd]
  bf16_t* Kb   = Qb  + (size_t)M_TOTAL * D_MODEL;      // [B,H,S,hd]
  bf16_t* Vtb  = Kb  + (size_t)M_TOTAL * D_MODEL;      // [B,H,hd,S]
  bf16_t* ctxb = Vtb + (size_t)M_TOTAL * D_MODEL;      // [B,S,D]

  const int nx4 = (M_TOTAL * D_MODEL) / 4;             // 1048576
  cvt_f32_to_bf16<<<nx4 / 256, 256, 0, stream>>>(x, xb, nx4);
  transpose_cvt_kernel<<<dim3(32, 32), 256, 0, stream>>>(Wq, wqt);
  transpose_cvt_kernel<<<dim3(32, 32), 256, 0, stream>>>(Wk, wkt);
  transpose_cvt_kernel<<<dim3(32, 32), 256, 0, stream>>>(Wv, wvt);
  transpose_cvt_kernel<<<dim3(32, 32), 256, 0, stream>>>(Wo, wot);

  gemm_qkv_kernel<<<dim3(M_TOTAL / 128, D_MODEL / 128, 3), 256, 0, stream>>>(
      xb, wqt, wkt, wvt, Qb, Kb, Vtb);

  flash_attn_kernel<<<dim3(SEQ / 64, BATCH * N_HEADS), 128, 0, stream>>>(
      Qb, Kb, Vtb, ctxb);

  gemm_out_kernel<<<dim3(M_TOTAL / 128, D_MODEL / 128), 256, 0, stream>>>(
      ctxb, wot, bo, (float*)d_out);
}
